// SpatialAttention_36601711297023
// MI455X (gfx1250) — matrix operand order repaired and checked by reading, not compiled, not run on hardware
//
#include <hip/hip_runtime.h>

typedef __attribute__((ext_vector_type(16))) _Float16 v16h;
typedef __attribute__((ext_vector_type(8)))  float    v8f;

#if __has_builtin(__builtin_amdgcn_tensor_load_to_lds) && \
    __has_builtin(__builtin_amdgcn_s_wait_tensorcnt)
#define USE_TDM 1
#else
#define USE_TDM 0
#endif

namespace {

constexpr int kB = 32, kT = 12, kN = 325, kD = 64, k2D = 128;
constexpr int kH = 8, kHD = 8;
constexpr int kNT = 21;           // ceil(325/16) row tiles
constexpr int kNPad = 336;        // 21*16
constexpr int kPStride = 352;     // P-strip row stride (halfs), K=32 multiple
constexpr int kVStride = 352;     // transposed-V row stride (halfs)
constexpr int kWaves = 8;
constexpr float kQScale = 0.35355339059327373f;  // 1/sqrt(DIM_HEADS)

// LDS layout (bytes), dynamic shared base = offset 0
constexpr unsigned kOffQ    = 0;                                  // [336][64] f16
constexpr unsigned kOffK    = kOffQ + kNPad * kD * 2;             // [336][64] f16
constexpr unsigned kOffV    = kOffK + kNPad * kD * 2;             // [64][352] f16 (V^T)
constexpr unsigned kOffW    = kOffV + kD * kVStride * 2;          // Wq,Wk,Wv [64][128] + Wo [64][64] f16
constexpr unsigned kOffBias = kOffW + (3 * kD * k2D + kD * kD) * 2;
constexpr unsigned kOffXc   = kOffBias + 4 * kD * 4;              // [8][16][128] f16
constexpr unsigned kOffP    = kOffXc + kWaves * 16 * k2D * 2;     // [8][16][352] f16
constexpr unsigned kStripBytes = 16 * kPStride * 2;               // 11264 per wave
constexpr size_t   kSmemBytes  = kOffP + kWaves * kStripBytes;    // 312320 < 320 KB

union Frag16 { v16h h; unsigned int u[8]; };

__device__ inline unsigned int ld32(const _Float16* p) {
  return *reinterpret_cast<const unsigned int*>(p);
}

// Dense A fragment (16x32 f16) from row-major f16 [rows][stride].
__device__ inline v16h loadA(const _Float16* base, int stride, int mr, int hf, int k0) {
  Frag16 f;
#pragma unroll
  for (int v = 0; v < 8; ++v) {
    int kl = hf * 16 + 2 * v;
    f.u[v] = ld32(base + mr * stride + k0 + kl);
  }
  return f.h;
}

// Dense B fragment (32x16 f16) from n-major f16 [n][stride] (K contiguous per column n).
__device__ inline v16h loadBT(const _Float16* base, int stride, int n, int hf, int k0) {
  Frag16 f;
#pragma unroll
  for (int v = 0; v < 8; ++v) f.u[v] = ld32(base + n * stride + k0 + hf * 16 + 2 * v);
  return f.h;
}

__device__ inline v8f wmma_f16(v16h a, v16h b, v8f c) {
  return __builtin_amdgcn_wmma_f32_16x16x32_f16(false, a, false, b, (short)0, c, false, false);
}

__device__ inline float redmax16(float v) {
  v = fmaxf(v, __shfl_xor(v, 1, 32));
  v = fmaxf(v, __shfl_xor(v, 2, 32));
  v = fmaxf(v, __shfl_xor(v, 4, 32));
  v = fmaxf(v, __shfl_xor(v, 8, 32));
  return v;
}
__device__ inline float redsum16(float v) {
  v += __shfl_xor(v, 1, 32);
  v += __shfl_xor(v, 2, 32);
  v += __shfl_xor(v, 4, 32);
  v += __shfl_xor(v, 8, 32);
  return v;
}

#if USE_TDM
typedef __attribute__((ext_vector_type(4))) unsigned int u32x4;
typedef __attribute__((ext_vector_type(8))) int i32x8;
typedef __attribute__((ext_vector_type(4))) int i32x4;

// TDM 2D tile load: 16 rows x 64 f32, row stride 64; rows beyond rowsRemaining read 0.
__device__ inline void tdm_load_tile_f32(unsigned ldsOff, const float* g, int rowsRemaining) {
  unsigned long long ga = (unsigned long long)(size_t)g;
  u32x4 g0 = { 1u,                                    // count=1 (valid user D#)
               ldsOff,                                // lds_addr
               (unsigned)(ga & 0xffffffffull),        // global_addr[31:0]
               (unsigned)((ga >> 32) & 0x1ffffffull)  // global_addr[56:32]
                 | (2u << 30) };                      // type=2 ("image")
  i32x8 g1 = { (int)(2u << 16),                       // data_size=4B
               (int)(64u << 16),                      // tensor_dim0 = 64
               (int)((unsigned)rowsRemaining << 16),  // tensor_dim1 = rows remaining
               (int)(64u << 16),                      // tile_dim0 = 64
               16,                                    // tile_dim1 = 16, tile_dim2 = 0
               64,                                    // tensor_dim0_stride = 64
               0, 0 };
  i32x4 gz = { 0, 0, 0, 0 };
#if __clang_major__ >= 23
  i32x8 gz8 = { 0, 0, 0, 0, 0, 0, 0, 0 };
  __builtin_amdgcn_tensor_load_to_lds(g0, g1, gz, gz, gz8, 0);
#else
  __builtin_amdgcn_tensor_load_to_lds(g0, g1, gz, gz, 0);
#endif
}
#endif  // USE_TDM

} // namespace

__global__ __launch_bounds__(256)
void spatial_attn_kernel(const float* __restrict__ X, const float* __restrict__ STE,
                         const float* __restrict__ Wq, const float* __restrict__ bq,
                         const float* __restrict__ Wk, const float* __restrict__ bk,
                         const float* __restrict__ Wv, const float* __restrict__ bv,
                         const float* __restrict__ Wo, const float* __restrict__ bo,
                         float* __restrict__ Out) {
  extern __shared__ char smem_raw[];
  _Float16* sQ  = reinterpret_cast<_Float16*>(smem_raw + kOffQ);   // reused as ctx
  _Float16* sK  = reinterpret_cast<_Float16*>(smem_raw + kOffK);
  _Float16* sVt = reinterpret_cast<_Float16*>(smem_raw + kOffV);
  _Float16* sWq = reinterpret_cast<_Float16*>(smem_raw + kOffW);
  _Float16* sWk = sWq + kD * k2D;
  _Float16* sWv = sWk + kD * k2D;
  _Float16* sWo = sWv + kD * k2D;
  float*    sBias = reinterpret_cast<float*>(smem_raw + kOffBias);
  _Float16* sXc = reinterpret_cast<_Float16*>(smem_raw + kOffXc);
  _Float16* sP  = reinterpret_cast<_Float16*>(smem_raw + kOffP);

  const int tid  = threadIdx.x;
  const int wv   = tid >> 5;
  const int lane = tid & 31;
  const int nc   = lane & 15;
  const int hf   = lane >> 4;

  const size_t btOff = (size_t)blockIdx.x * kN * kD;
  const float* Xbt = X + btOff;
  const float* Sbt = STE + btOff;
  float* Obt = Out + btOff;

  _Float16* myXc = sXc + wv * 16 * k2D;
  _Float16* myP  = sP + wv * 16 * kPStride;

  // ---------------- Phase 0: weights (f16, transposed) + biases -> LDS ----------------
  for (int i = tid; i < k2D * kD; i += 256) {
    int k = i >> 6, n = i & 63;
    sWq[n * k2D + k] = (_Float16)Wq[i];
    sWk[n * k2D + k] = (_Float16)Wk[i];
    sWv[n * k2D + k] = (_Float16)Wv[i];
  }
  for (int i = tid; i < kD * kD; i += 256) {
    int k = i >> 6, n = i & 63;
    sWo[n * kD + k] = (_Float16)Wo[i];
  }
  if (tid < kD) {
    sBias[tid]          = bq[tid];
    sBias[kD + tid]     = bk[tid];
    sBias[2 * kD + tid] = bv[tid];
    sBias[3 * kD + tid] = bo[tid];
  }
  for (int i = tid; i < kD * 16; i += 256) {              // zero V pad keys 336..351
    sVt[(i >> 4) * kVStride + kNPad + (i & 15)] = (_Float16)0.f;
  }
  __syncthreads();

  // ---------------- Phase 1: Q/K/V projections, per-wave row tiles ----------------
  for (int rt = wv; rt < kNT; rt += kWaves) {
#if USE_TDM
    {
      // TDM: X and STE 16x64 f32 tiles -> f32 stage (overlays this wave's P strip).
      // OOB rows (>= kN) read back as zero, so no masking needed.
      unsigned stOff = kOffP + (unsigned)wv * kStripBytes;
      tdm_load_tile_f32(stOff,        Xbt + rt * 16 * kD, kN - rt * 16);
      tdm_load_tile_f32(stOff + 4096, Sbt + rt * 16 * kD, kN - rt * 16);
      __builtin_amdgcn_s_wait_tensorcnt(0);
      asm volatile("" ::: "memory");
      const float* stX = reinterpret_cast<const float*>(myP);
      const float* stS = stX + 16 * kD;
      for (int i = lane; i < 16 * k2D; i += 32) {
        int r = i >> 7, c = i & 127;
        myXc[i] = (_Float16)((c < kD) ? stX[r * kD + c] : stS[r * kD + (c - kD)]);
      }
    }
#else
    for (int i = lane; i < 16 * k2D; i += 32) {
      int r = i >> 7, c = i & 127;
      int grow = rt * 16 + r;
      float val = 0.f;
      if (grow < kN) val = (c < kD) ? Xbt[grow * kD + c] : Sbt[grow * kD + (c - kD)];
      myXc[i] = (_Float16)val;
    }
#endif
    asm volatile("s_wait_dscnt 0x0" ::: "memory");

    // Hoist the 4 Xc A-fragments (reused by all 3 projections x 4 n-tiles).
    v16h afr[4];
#pragma unroll
    for (int kk = 0; kk < 4; ++kk) afr[kk] = loadA(myXc, k2D, nc, hf, kk * 32);

#pragma unroll
    for (int proj = 0; proj < 3; ++proj) {   // fully unrolled: static store targets
      const _Float16* W = (proj == 0) ? sWq : ((proj == 1) ? sWk : sWv);
      const float* bias = sBias + proj * kD;
#pragma unroll 1
      for (int nt = 0; nt < 4; ++nt) {
        v8f acc = {};
#pragma unroll
        for (int kk = 0; kk < 4; ++kk)
          acc = wmma_f16(afr[kk], loadBT(W, k2D, nt * 16 + nc, hf, kk * 32), acc);
        float bsc = bias[nt * 16 + nc];
#pragma unroll
        for (int i = 0; i < 8; ++i) {
          int row = rt * 16 + i + 8 * hf;
          int col = nt * 16 + nc;
          float val = fmaxf(acc[i] + bsc, 0.f);
          if (proj == 0)      sQ[row * kD + col] = (_Float16)(val * kQScale);
          else if (proj == 1) sK[row * kD + col] = (_Float16)val;
          else                sVt[col * kVStride + row] = (_Float16)val;
        }
      }
    }
  }
  __syncthreads();

  // ---------------- Phase 2: attention + output projection ----------------
  for (int qt = wv; qt < kNT; qt += kWaves) {
    const _Float16* qBase = sQ + qt * 16 * kD;
#pragma unroll 1
    for (int h = 0; h < kH; ++h) {
      // ---- S = (Q/sqrt(d)) K^T : head dim 8 zero-padded into K=32 ----
      Frag16 fa;
#pragma unroll
      for (int v = 0; v < 8; ++v)
        fa.u[v] = (hf == 0 && v < 4) ? ld32(qBase + nc * kD + h * kHD + 2 * v) : 0u;

      float tmax[8];
#pragma unroll
      for (int i = 0; i < 8; ++i) tmax[i] = -3.0e30f;

#pragma unroll 1
      for (int kt = 0; kt < kNT; ++kt) {
        Frag16 fb;
#pragma unroll
        for (int v = 0; v < 8; ++v)
          fb.u[v] = (hf == 0 && v < 4) ? ld32(sK + (kt * 16 + nc) * kD + h * kHD + 2 * v) : 0u;
        v8f s = {};
        s = wmma_f16(fa.h, fb.h, s);
        int key = kt * 16 + nc;
        bool kvalid = key < kN;
#pragma unroll
        for (int i = 0; i < 8; ++i) {
          float sv = kvalid ? s[i] : -60000.f;
          myP[(i + 8 * hf) * kPStride + key] = (_Float16)sv;
          tmax[i] = fmaxf(tmax[i], sv);
        }
      }
      float rmax[8], rsum[8];
#pragma unroll
      for (int i = 0; i < 8; ++i) { rmax[i] = redmax16(tmax[i]); rsum[i] = 0.f; }
      asm volatile("s_wait_dscnt 0x0" ::: "memory");

      // ---- softmax pass 2: exp + row sums, P stored back f16 ----
#pragma unroll 1
      for (int kt = 0; kt < kNT; ++kt) {
        int key = kt * 16 + nc;
#pragma unroll
        for (int i = 0; i < 8; ++i) {
          float sv = (float)myP[(i + 8 * hf) * kPStride + key];
          float e = (key < kN) ? __expf(sv - rmax[i]) : 0.f;
          myP[(i + 8 * hf) * kPStride + key] = (_Float16)e;
          rsum[i] += e;
        }
      }
#pragma unroll
      for (int i = 0; i < 8; ++i)  // zero strip pad cols 336..351 (K=32 multiple)
        myP[(i + 8 * hf) * kPStride + kNPad + nc] = (_Float16)0.f;
      float rinv[8];
#pragma unroll
      for (int i = 0; i < 8; ++i) rinv[i] = 1.f / redsum16(rsum[i]);
      asm volatile("s_wait_dscnt 0x0" ::: "memory");

      // ---- ctx_h = P V_h : K over 352 padded keys, N cols 0..7 valid ----
      v8f ctx = {};
#pragma unroll 1
      for (int kk = 0; kk < 11; ++kk) {
        v16h a = loadA(myP, kPStride, nc, hf, kk * 32);
        Frag16 fb;
#pragma unroll
        for (int v = 0; v < 8; ++v)
          fb.u[v] = (nc < kHD)
                  ? ld32(sVt + (h * kHD + nc) * kVStride + kk * 32 + hf * 16 + 2 * v) : 0u;
        ctx = wmma_f16(a, fb.h, ctx);
      }
      if (nc < kHD) {
#pragma unroll
        for (int i = 0; i < 8; ++i) {   // normalized ctx overwrites this head's Q columns
          int row = qt * 16 + i + 8 * hf;
          sQ[row * kD + h * kHD + nc] = (_Float16)(ctx[i] * rinv[i]);
        }
      }
      asm volatile("s_wait_dscnt 0x0" ::: "memory");
    }

    // ---- out = relu(ctx @ W_o + b_o) ----
    v16h ca[2];
#pragma unroll
    for (int kk = 0; kk < 2; ++kk) ca[kk] = loadA(qBase, kD, nc, hf, kk * 32);
#pragma unroll 1
    for (int nt = 0; nt < 4; ++nt) {
      v8f acc = {};
#pragma unroll
      for (int kk = 0; kk < 2; ++kk)
        acc = wmma_f16(ca[kk], loadBT(sWo, kD, nt * 16 + nc, hf, kk * 32), acc);
      float bsc = sBias[3 * kD + nt * 16 + nc];
#pragma unroll
      for (int i = 0; i < 8; ++i) {
        int row = qt * 16 + i + 8 * hf;
        if (row < kN) Obt[row * kD + nt * 16 + nc] = fmaxf(acc[i] + bsc, 0.f);
      }
    }
  }
}

extern "C" void kernel_launch(void* const* d_in, const int* in_sizes, int n_in,
                              void* d_out, int out_size, void* d_ws, size_t ws_size,
                              hipStream_t stream) {
  (void)in_sizes; (void)n_in; (void)d_ws; (void)ws_size; (void)out_size;
  const float* X   = (const float*)d_in[0];
  const float* STE = (const float*)d_in[1];
  const float* Wq  = (const float*)d_in[2];
  const float* bq  = (const float*)d_in[3];
  const float* Wk  = (const float*)d_in[4];
  const float* bk  = (const float*)d_in[5];
  const float* Wv  = (const float*)d_in[6];
  const float* bv  = (const float*)d_in[7];
  const float* Wo  = (const float*)d_in[8];
  const float* bo  = (const float*)d_in[9];
  float* Out = (float*)d_out;

  dim3 grid(kB * kT);   // 384 blocks: one (b,t) per WGP
  dim3 block(256);      // 8 wave32s
  spatial_attn_kernel<<<grid, block, kSmemBytes, stream>>>(
      X, STE, Wq, bq, Wk, bk, Wv, bv, Wo, bo, Out);
}
